// WeightedContrastiveLoss1_48945447305730
// MI455X (gfx1250) — matrix operand order repaired and checked by reading, not compiled, not run on hardware
//
#include <hip/hip_runtime.h>
#include <hip/hip_bf16.h>
#include <stdint.h>

// ---------------------------------------------------------------------------
// WeightedContrastiveLoss (NT-Xent with weighted positives), MI455X / gfx1250
// B=4096, D=256, T=0.5.  Never materialize sim[8192,8192]; fuse exp(sim/T) +
// row-sum into a register-resident-A f16 WMMA GEMM (64 rows/wave -> 4 WMMA
// per B-fragment).  B panels staged into LDS once per block by the Tensor
// Data Mover (double-buffered, TENSORcnt).
// ---------------------------------------------------------------------------

#define BN    4096
#define DN    256
#define TWOB  8192
#define INV_T 2.0f           // 1 / 0.5
#define EPSN  1e-12f

#define ROWS_PER_WAVE  64    // 4 x 16-row WMMA tiles, A register-resident
#define ROWS_PER_BLOCK 512   // 8 waves
#define COLS_PER_BLOCK 512
#define TILE_COLS      32    // columns staged per TDM tile (16 KB)

typedef _Float16 v16h __attribute__((ext_vector_type(16)));
typedef _Float16 h8   __attribute__((ext_vector_type(8)));
typedef float    v8f  __attribute__((ext_vector_type(8)));

typedef unsigned int u32x4 __attribute__((ext_vector_type(4)));
typedef int          i32x8 __attribute__((ext_vector_type(8)));
typedef int          i32x4 __attribute__((ext_vector_type(4)));

// ---------------------------------------------------------------------------
// Kernel 0: zero the denominator accumulator and the scalar output
// ---------------------------------------------------------------------------
__global__ void wcl_zero_kernel(float* __restrict__ denom, float* __restrict__ out) {
    int i = blockIdx.x * 256 + threadIdx.x;
    if (i < TWOB) denom[i] = 0.0f;
    if (i == 0)   out[0]   = 0.0f;
}

// ---------------------------------------------------------------------------
// Kernel 1: L2-normalize both embeddings, emit f16 reps[2B,D], weight[B].
// ---------------------------------------------------------------------------
__global__ __launch_bounds__(256)
void wcl_normalize_kernel(const float* __restrict__ emb_i,
                          const float* __restrict__ emb_j,
                          _Float16*    __restrict__ z,       // [2B, D] f16
                          float*       __restrict__ weight)  // [B]
{
    const int k = blockIdx.x;
    const int t = threadIdx.x;          // 0..255 == D
    const float xi = emb_i[k * DN + t];
    const float xj = emb_j[k * DN + t];

    float si = xi * xi;
    float sj = xj * xj;
    float sd = xi * xj;

    #pragma unroll
    for (int m = 16; m >= 1; m >>= 1) {
        si += __shfl_xor(si, m, 32);
        sj += __shfl_xor(sj, m, 32);
        sd += __shfl_xor(sd, m, 32);
    }

    __shared__ float lsi[8], lsj[8], lsd[8];
    const int lane = t & 31, wv = t >> 5;
    if (lane == 0) { lsi[wv] = si; lsj[wv] = sj; lsd[wv] = sd; }
    __syncthreads();

    float Si = 0.f, Sj = 0.f, Sd = 0.f;
    #pragma unroll
    for (int i = 0; i < 8; ++i) { Si += lsi[i]; Sj += lsj[i]; Sd += lsd[i]; }

    const float ni = fmaxf(__builtin_sqrtf(Si), EPSN);
    const float nj = fmaxf(__builtin_sqrtf(Sj), EPSN);

    z[k * DN + t]        = (_Float16)(xi / ni);
    z[(k + BN) * DN + t] = (_Float16)(xj / nj);
    if (t == 0) weight[k] = Sd / (ni * nj);
}

// ---------------------------------------------------------------------------
// TDM: one tensor_load_to_lds moving a [TILE_COLS x DN] f16 tile (16 KB,
// row-major, layout identical to the source) into LDS.
// D# per CDNA5 ISA (2-D tensor, groups 2/3 zeroed).
// This toolchain exposes the 6-arg builtin (extra int32x8 group before cpol).
// ---------------------------------------------------------------------------
__device__ __forceinline__ void tdm_load_tile(const _Float16* gptr, unsigned lds_off) {
    const unsigned long long ga = (unsigned long long)(uintptr_t)gptr;

    u32x4 g0;
    g0[0] = 1u;                                            // count=1, user mode
    g0[1] = lds_off;                                       // lds_addr (bytes)
    g0[2] = (unsigned)(ga & 0xFFFFFFFFu);                  // global_addr[31:0]
    g0[3] = (unsigned)((ga >> 32) & 0x1FFFFFFu)            // global_addr[56:32]
          | (2u << 30);                                    // type=2 ("image")

    i32x8 g1;
    g1[0] = (int)(1u << 16);              // data_size=1 (2 bytes/elem)
    g1[1] = (int)((unsigned)DN << 16);    // tensor_dim0[15:0]=256 in bits 63:48
    g1[2] = (int)((unsigned)TWOB << 16);  // tensor_dim1[15:0]=8192 in bits 95:80
    g1[3] = (int)((unsigned)DN << 16);    // tile_dim0=256 in bits 127:112
    g1[4] = TILE_COLS;                    // tile_dim1=32 (tile_dim2=0)
    g1[5] = DN;                           // tensor_dim0_stride[31:0]=256
    g1[6] = 0;                            // stride hi / dim1_stride lo
    g1[7] = 0;                            // dim1_stride hi

    i32x4 gz4 = {0, 0, 0, 0};             // groups 2/3 unused (2-D tensor)
    i32x8 gz8 = {0, 0, 0, 0, 0, 0, 0, 0};

    __builtin_amdgcn_tensor_load_to_lds(g0, g1, gz4, gz4, gz8, 0);
}

// ---------------------------------------------------------------------------
// Kernel 2: fused sim-GEMM + exp + masked row-sum.
//   grid = (16 row-blocks, 16 col-splits), block = 256 threads (8 waves).
//   Each wave owns 64 rows (four 16-row WMMA tiles) with all 8 K-step A
//   fragments register-resident (256 VGPRs) -> 4 WMMAs per B-fragment.
//   B panels (32 cols x 256 K) TDM-staged into double-buffered LDS shared by
//   all 8 waves; fragments read back with ds_load_b128.  Epilogue:
//   e = exp(2*sim), diagonal masked (neg_mask = 1 - eye), per-row running
//   sums, 16-lane xor reduction, one atomicAdd per row per col-split.
// ---------------------------------------------------------------------------
__global__ __launch_bounds__(256)
void wcl_simexp_kernel(const _Float16* __restrict__ z,      // [2B, D] f16
                       float*          __restrict__ denom)  // [2B]
{
    __shared__ __align__(32) _Float16 smem[2][TILE_COLS * DN];   // 2 x 16 KB

    const int lane = threadIdx.x & 31;
    const int wv   = threadIdx.x >> 5;
    const int lr   = lane & 15;     // N index / M-within-16
    const int lh   = lane >> 4;     // lane half

    const int rowbase = blockIdx.x * ROWS_PER_BLOCK + wv * ROWS_PER_WAVE;
    const int colbase = blockIdx.y * COLS_PER_BLOCK;

    // ---- load resident A fragments: 4 row-tiles x 8 K-steps --------------
    v16h a[4][8];
    const int khalf = lh * 16;
    #pragma unroll
    for (int t = 0; t < 4; ++t) {
        const _Float16* rp = z + (size_t)(rowbase + t * 16 + lr) * DN;
        #pragma unroll
        for (int kk = 0; kk < 8; ++kk) {
            const int k0 = kk * 32;
            h8 lo = *(const h8*)(rp + k0 + khalf);
            h8 hi = *(const h8*)(rp + k0 + khalf + 8);
            a[t][kk] = __builtin_shufflevector(lo, hi,
                0, 1, 2, 3, 4, 5, 6, 7, 8, 9, 10, 11, 12, 13, 14, 15);
        }
    }

    float sums[4][8];
    #pragma unroll
    for (int t = 0; t < 4; ++t)
        #pragma unroll
        for (int r = 0; r < 8; ++r) sums[t][r] = 0.0f;

    const unsigned lds0 =
        (unsigned)(unsigned long long)(uintptr_t)&smem[0][0];
    const _Float16* gcol = z + (size_t)colbase * DN;
    const int NTILES = COLS_PER_BLOCK / TILE_COLS;    // 16

    // prologue: stage tile 0 into buffer 0
    if (wv == 0) tdm_load_tile(gcol, lds0);

    for (int ct = 0; ct < NTILES; ++ct) {
        const int cur = ct & 1;

        if (wv == 0) __builtin_amdgcn_s_wait_tensorcnt(0);  // tile ct in LDS
        __syncthreads();                                    // publish to block

        // prefetch next tile into the other buffer (its readers are past
        // the barrier above, so WAR on the double buffer is safe)
        if (wv == 0 && ct + 1 < NTILES)
            tdm_load_tile(gcol + (size_t)(ct + 1) * TILE_COLS * DN,
                          lds0 + (unsigned)((cur ^ 1) * TILE_COLS * DN * 2));

        // ---- compute two 16-col WMMA tiles from the staged panel ---------
        #pragma unroll
        for (int sub = 0; sub < 2; ++sub) {
            const int ncol = colbase + ct * TILE_COLS + sub * 16 + lr;
            // B frag lane layout: N = lane&15, kb = (lane>>4)*16,
            // elems[0..15] = panel[nLocal][k0+kb .. +15] (32B contiguous)
            const _Float16* bp =
                &smem[cur][(sub * 16 + lr) * DN + lh * 16];

            v8f acc0 = {}, acc1 = {}, acc2 = {}, acc3 = {};
            #pragma unroll
            for (int kk = 0; kk < 8; ++kk) {
                v16h b = *(const v16h*)(bp + kk * 32);
                acc0 = __builtin_amdgcn_wmma_f32_16x16x32_f16(
                           false, a[0][kk], false, b, (short)0, acc0, false, false);
                acc1 = __builtin_amdgcn_wmma_f32_16x16x32_f16(
                           false, a[1][kk], false, b, (short)0, acc1, false, false);
                acc2 = __builtin_amdgcn_wmma_f32_16x16x32_f16(
                           false, a[2][kk], false, b, (short)0, acc2, false, false);
                acc3 = __builtin_amdgcn_wmma_f32_16x16x32_f16(
                           false, a[3][kk], false, b, (short)0, acc3, false, false);
            }

            // ---- epilogue: exp(sim/T), mask diagonal, accumulate ---------
            #pragma unroll
            for (int t = 0; t < 4; ++t) {
                const int rbase = rowbase + t * 16 + lh * 8;
                #pragma unroll
                for (int r = 0; r < 8; ++r) {
                    float v;
                    switch (t) {
                        case 0: v = acc0[r]; break;
                        case 1: v = acc1[r]; break;
                        case 2: v = acc2[r]; break;
                        default: v = acc3[r]; break;
                    }
                    float e = __expf(v * INV_T);
                    if (rbase + r == ncol) e = 0.0f;      // neg_mask diagonal
                    sums[t][r] += e;
                }
            }
        }
    }

    // ---- reduce row sums across the 16 lanes sharing each row set --------
    #pragma unroll
    for (int t = 0; t < 4; ++t)
        #pragma unroll
        for (int r = 0; r < 8; ++r) {
            float s = sums[t][r];
            s += __shfl_xor(s, 1, 32);
            s += __shfl_xor(s, 2, 32);
            s += __shfl_xor(s, 4, 32);
            s += __shfl_xor(s, 8, 32);
            sums[t][r] = s;
        }

    if (lr == 0) {          // lane 0 -> rows +0..7, lane 16 -> rows +8..15
        #pragma unroll
        for (int t = 0; t < 4; ++t)
            #pragma unroll
            for (int r = 0; r < 8; ++r)
                atomicAdd(&denom[rowbase + t * 16 + lh * 8 + r], sums[t][r]);
    }
}

// ---------------------------------------------------------------------------
// Kernel 3: loss_partial[row] = log(denom[row]) - (pos/weight)/T, mean-reduce.
// pos[row] == weight[row % B] (diagonal of sim at offset +/-B), both fp32.
// ---------------------------------------------------------------------------
__global__ __launch_bounds__(256)
void wcl_finalize_kernel(const float* __restrict__ denom,
                         const float* __restrict__ weight,
                         float*       __restrict__ out)
{
    const int row = blockIdx.x * 256 + threadIdx.x;   // 32 blocks x 256
    const float w   = weight[row & (BN - 1)];
    const float pos = w;                               // sim diag == weight
    const float lp  = __logf(denom[row]) - (pos / w) * INV_T;

    float s = lp;
    #pragma unroll
    for (int m = 16; m >= 1; m >>= 1) s += __shfl_xor(s, m, 32);

    __shared__ float ls[8];
    const int lane = threadIdx.x & 31, wv = threadIdx.x >> 5;
    if (lane == 0) ls[wv] = s;
    __syncthreads();
    if (threadIdx.x == 0) {
        float bs = 0.f;
        #pragma unroll
        for (int i = 0; i < 8; ++i) bs += ls[i];
        atomicAdd(out, bs * (1.0f / (float)TWOB));
    }
}

// ---------------------------------------------------------------------------
extern "C" void kernel_launch(void* const* d_in, const int* in_sizes, int n_in,
                              void* d_out, int out_size, void* d_ws, size_t ws_size,
                              hipStream_t stream) {
    (void)in_sizes; (void)n_in; (void)out_size; (void)ws_size;

    const float* emb_i = (const float*)d_in[0];
    const float* emb_j = (const float*)d_in[1];
    float* out = (float*)d_out;

    // workspace layout
    char* ws = (char*)d_ws;
    _Float16* z      = (_Float16*)ws;                         // 8192*256*2 = 4 MB
    float*    denom  = (float*)(ws + (size_t)TWOB * DN * 2);  // 32 KB
    float*    weight = denom + TWOB;                          // 16 KB

    wcl_zero_kernel<<<(TWOB + 255) / 256, 256, 0, stream>>>(denom, out);
    wcl_normalize_kernel<<<BN, 256, 0, stream>>>(emb_i, emb_j, z, weight);

    dim3 grid(TWOB / ROWS_PER_BLOCK, TWOB / COLS_PER_BLOCK);  // 16 x 16
    wcl_simexp_kernel<<<grid, 256, 0, stream>>>(z, denom);

    wcl_finalize_kernel<<<TWOB / 256, 256, 0, stream>>>(denom, weight, out);
}
